// ResidualSparseSkillMLP_44341242364489
// MI455X (gfx1250) — compile-verified, matrix operand order checked by reading, not run on hardware
//
#include <hip/hip_runtime.h>
#include <hip/hip_bf16.h>
#include <stdint.h>

typedef __attribute__((ext_vector_type(16))) __bf16       v16bf;
typedef __attribute__((ext_vector_type(8)))  float        v8f;
typedef __attribute__((ext_vector_type(4)))  unsigned int v4u;

#define T_TOK  16384
#define HDIM   1024
#define FDIM   1024
#define NEXP   32
#define CAP    640
#define NCHUNK 512   // T_TOK / 32

__device__ __forceinline__ unsigned short f2bf(float f) {
  unsigned u = __float_as_uint(f);
  unsigned r = (u + 0x7FFFu + ((u >> 16) & 1u)) >> 16;  // RNE
  return (unsigned short)r;
}

#if __has_builtin(__builtin_amdgcn_cvt_pk_bf16_f32)
typedef __attribute__((ext_vector_type(2))) __bf16 v2bf;
__device__ __forceinline__ unsigned pk2bf(float lo, float hi) {
  v2bf p = __builtin_amdgcn_cvt_pk_bf16_f32(lo, hi);   // v_cvt_pk_bf16_f32
  return __builtin_bit_cast(unsigned, p);
}
#else
// Builtin not declared on this toolchain, but CDNA5 carries packed-BF16 VALU
// (ISA §15.10), so emit the conversion directly.
__device__ __forceinline__ unsigned pk2bf(float lo, float hi) {
  unsigned r;
  asm("v_cvt_pk_bf16_f32 %0, %1, %2" : "=v"(r) : "v"(lo), "v"(hi));
  return r;
}
#endif

// ---- CDNA5 async global->LDS copy (ISA §10.6 / §15.18, tracked by ASYNCcnt)
__device__ __forceinline__ void async_ld_b128(unsigned lds_off, const void* gaddr) {
  asm volatile("global_load_async_to_lds_b128 %0, %1, off"
               :: "v"(lds_off), "v"(gaddr) : "memory");
}
__device__ __forceinline__ void wait_async0() {
  asm volatile("s_wait_asynccnt 0x0" ::: "memory");
}
__device__ __forceinline__ unsigned lds_addr(const void* p) {
  return (unsigned)(size_t)p;   // low 32 bits of generic ptr = LDS offset
}

// ---------------- 1. router: logits, softmax max/sum, argmax ----------------
__global__ __launch_bounds__(256) void moe_router(const float* __restrict__ x,
                                                  const float* __restrict__ wr,
                                                  int* __restrict__ top1,
                                                  float* __restrict__ gate) {
  int wave = threadIdx.x >> 5;
  int lane = threadIdx.x & 31;
  int t = blockIdx.x * 8 + wave;
  const float* xr = x + (size_t)t * HDIM;
  float acc = 0.f;
  for (int h0 = 0; h0 < HDIM; h0 += 4) {
    float4 xv = *(const float4*)(xr + h0);        // broadcast across lanes
    acc += xv.x * wr[(h0 + 0) * NEXP + lane];
    acc += xv.y * wr[(h0 + 1) * NEXP + lane];
    acc += xv.z * wr[(h0 + 2) * NEXP + lane];
    acc += xv.w * wr[(h0 + 3) * NEXP + lane];
  }
  float m = acc; int mi = lane;                   // (max logit, lowest index)
  for (int d = 1; d < 32; d <<= 1) {
    float om = __shfl_xor(m, d, 32);
    int   oi = __shfl_xor(mi, d, 32);
    if (om > m || (om == m && oi < mi)) { m = om; mi = oi; }
  }
  float s = __expf(acc - m);
  for (int d = 1; d < 32; d <<= 1) s += __shfl_xor(s, d, 32);
  if (lane == 0) { top1[t] = mi; gate[t] = 1.0f / s; }  // prob at argmax
}

// ------------- 2. per-chunk expert counts + in-chunk ranks (ballot) ---------
__global__ __launch_bounds__(256) void moe_count(const int* __restrict__ top1,
                                                 int* __restrict__ chunkCnt,
                                                 int* __restrict__ rank) {
  int wave = threadIdx.x >> 5, lane = threadIdx.x & 31;
  int chunk = blockIdx.x * 8 + wave;
  int t = chunk * 32 + lane;
  int e = top1[t];
  int myrank = 0, cnt = 0;
  #pragma unroll
  for (int ee = 0; ee < NEXP; ++ee) {
    unsigned mask = __builtin_amdgcn_ballot_w32(e == ee);
    if (lane == ee) cnt = __popc(mask);
    if (e == ee)    myrank = __popc(mask & ((1u << lane) - 1u));
  }
  chunkCnt[chunk * 32 + lane] = cnt;
  rank[t] = myrank;
}

// ------------- 3. exclusive scan over chunks, per expert (1 wave) -----------
__global__ void moe_scan(const int* __restrict__ chunkCnt, int* __restrict__ chunkBase) {
  int lane = threadIdx.x;   // 32 threads, lane == expert
  int run = 0;
  for (int c = 0; c < NCHUNK; ++c) {
    int v = chunkCnt[c * 32 + lane];
    chunkBase[c * 32 + lane] = run;
    run += v;
  }
}

// ------------- 4. slot2tok init (must be -1 every call) ---------------------
__global__ void moe_init_s2t(int* __restrict__ p) {
  p[blockIdx.x * 256 + threadIdx.x] = -1;
}

// ------------- 5. final position / slot assignment --------------------------
__global__ __launch_bounds__(256) void moe_assign(const int* __restrict__ top1,
                                                  const int* __restrict__ rank,
                                                  const int* __restrict__ chunkBase,
                                                  int* __restrict__ slot,
                                                  int* __restrict__ slot2tok) {
  int t = blockIdx.x * 256 + threadIdx.x;
  int e = top1[t];
  int pos = chunkBase[(t >> 5) * 32 + e] + rank[t];
  if (pos < CAP) {
    int s = e * CAP + pos;
    slot[t] = s;
    slot2tok[s] = t;
  } else {
    slot[t] = -1;
  }
}

// ------------- 6. scatter to bf16 expert buffer + out = x (residual) --------
__global__ __launch_bounds__(256) void moe_scatter(const float* __restrict__ x,
                                                   const int* __restrict__ slot,
                                                   unsigned short* __restrict__ bufbf,
                                                   float* __restrict__ out) {
  int t = blockIdx.x;
  int h0 = threadIdx.x * 4;
  float4 xv = *(const float4*)(x + (size_t)t * HDIM + h0);
  *(float4*)(out + (size_t)t * HDIM + h0) = xv;   // residual / dropped path
  int s = slot[t];
  if (s >= 0) {
    uint2 pk;
    pk.x = pk2bf(xv.x, xv.y);
    pk.y = pk2bf(xv.z, xv.w);
    *(uint2*)(bufbf + (size_t)s * HDIM + h0) = pk;
  }
}

// ------------- WMMA fragment loader (A and transposed-B share layout) -------
// 16-bit A 16x32 layout: lane L<16 -> row M=L, ks {0..7}U{16..23};
// lane L+16 -> row M=L, ks {8..15}U{24..31}. Two 16B LDS chunks per lane.
union Frag { v4u q[2]; v16bf v; };
__device__ __forceinline__ v16bf load_frag(const unsigned short* base, int row, int hv) {
  Frag f;
  f.q[0] = *(const v4u*)(base + row * 40 + hv * 8);
  f.q[1] = *(const v4u*)(base + row * 40 + 16 + hv * 8);
  return f.v;
}

// Stage a 32(K) x 128(N) f32 tile into LDS transposed [n][k] as bf16.
// Each thread: k-pair (2*kp, 2*kp+1) x 8 consecutive n -> 8 packed b32 stores.
__device__ __forceinline__ void stage_b(unsigned short* sB, const float* src,
                                        int ld, int kp, int nc0) {
  const float* s0 = src + (size_t)(2 * kp) * ld + nc0;
  const float* s1 = s0 + ld;
  float4 a0 = *(const float4*)(s0);
  float4 a1 = *(const float4*)(s0 + 4);
  float4 b0 = *(const float4*)(s1);
  float4 b1 = *(const float4*)(s1 + 4);
  *(unsigned*)(sB + (size_t)(nc0 + 0) * 40 + 2 * kp) = pk2bf(a0.x, b0.x);
  *(unsigned*)(sB + (size_t)(nc0 + 1) * 40 + 2 * kp) = pk2bf(a0.y, b0.y);
  *(unsigned*)(sB + (size_t)(nc0 + 2) * 40 + 2 * kp) = pk2bf(a0.z, b0.z);
  *(unsigned*)(sB + (size_t)(nc0 + 3) * 40 + 2 * kp) = pk2bf(a0.w, b0.w);
  *(unsigned*)(sB + (size_t)(nc0 + 4) * 40 + 2 * kp) = pk2bf(a1.x, b1.x);
  *(unsigned*)(sB + (size_t)(nc0 + 5) * 40 + 2 * kp) = pk2bf(a1.y, b1.y);
  *(unsigned*)(sB + (size_t)(nc0 + 6) * 40 + 2 * kp) = pk2bf(a1.z, b1.z);
  *(unsigned*)(sB + (size_t)(nc0 + 7) * 40 + 2 * kp) = pk2bf(a1.w, b1.w);
}

// ------------- 7. fused gate+up GEMM + SwiGLU -> act (bf16) -----------------
// per block: 64(M) x 128(N) tile, K-tile 32, 8 waves as 4(M) x 2(N).
// Double-buffered LDS: stage tile k+1 before computing tile k.
__global__ __launch_bounds__(256) void moe_gemm1(const unsigned short* __restrict__ bufbf,
                                                 const float* __restrict__ wg,
                                                 const float* __restrict__ wu,
                                                 unsigned short* __restrict__ act) {
  __shared__ __align__(16) unsigned short sA [2][64  * 40];
  __shared__ __align__(16) unsigned short sBg[2][128 * 40];
  __shared__ __align__(16) unsigned short sBu[2][128 * 40];
  int bx = blockIdx.x;
  int e = bx / 80, rem = bx % 80;
  int m0 = (rem / 8) * 64, n0 = (rem % 8) * 128;
  int tid = threadIdx.x, lane = tid & 31, wave = tid >> 5;
  int wm = (wave & 3) * 16, wn = (wave >> 2) * 64;
  int hv = lane >> 4, l16 = lane & 15;

  v8f cg[4] = {}; v8f cu[4] = {};

  const unsigned short* Abase = bufbf + ((size_t)e * CAP + m0) * HDIM;
  const float* Bg = wg + (size_t)e * HDIM * FDIM + n0;
  const float* Bu = wu + (size_t)e * HDIM * FDIM + n0;
  int arow = tid >> 2, aseg = (tid & 3) * 8;     // A: 64 rows x 32 bf16
  int kp = tid >> 4, nc0 = (tid & 15) * 8;       // B: k-pair x 8n

  // prologue: stage k-tile 0 into buffer 0
  async_ld_b128(lds_addr(&sA[0][arow * 40 + aseg]),
                Abase + (size_t)arow * HDIM + aseg);
  stage_b(sBg[0], Bg, FDIM, kp, nc0);
  stage_b(sBu[0], Bu, FDIM, kp, nc0);

  int cur = 0;
  for (int k0 = 0; k0 < HDIM; k0 += 32) {
    wait_async0();        // our async write into sA[cur] (issued last iter)
    __syncthreads();      // all waves' stores to buffer `cur` visible
    int nxt = cur ^ 1;
    if (k0 + 32 < HDIM) { // stage next tile while computing this one
      async_ld_b128(lds_addr(&sA[nxt][arow * 40 + aseg]),
                    Abase + (size_t)arow * HDIM + (k0 + 32) + aseg);
      stage_b(sBg[nxt], Bg + (size_t)(k0 + 32) * FDIM, FDIM, kp, nc0);
      stage_b(sBu[nxt], Bu + (size_t)(k0 + 32) * FDIM, FDIM, kp, nc0);
      if (k0 + 96 <= HDIM) {
        __builtin_prefetch(Bg + (size_t)(k0 + 64 + 2 * kp) * FDIM + nc0, 0, 1);
        __builtin_prefetch(Bu + (size_t)(k0 + 64 + 2 * kp) * FDIM + nc0, 0, 1);
      }
    }
    // preload all fragments, then back-to-back WMMAs
    v16bf a = load_frag(sA[cur], wm + l16, hv);
    v16bf fg[4], fu[4];
    #pragma unroll
    for (int nt = 0; nt < 4; ++nt) {
      fg[nt] = load_frag(sBg[cur], wn + nt * 16 + l16, hv);
      fu[nt] = load_frag(sBu[cur], wn + nt * 16 + l16, hv);
    }
    #pragma unroll
    for (int nt = 0; nt < 4; ++nt) {
      cg[nt] = __builtin_amdgcn_wmma_f32_16x16x32_bf16(false, a, false, fg[nt],
                                                       (short)0, cg[nt], false, false);
      cu[nt] = __builtin_amdgcn_wmma_f32_16x16x32_bf16(false, a, false, fu[nt],
                                                       (short)0, cu[nt], false, false);
    }
    cur = nxt;
  }
  // epilogue: act = bf16(silu(g) * u), fast rcp instead of IEEE divide
  #pragma unroll
  for (int nt = 0; nt < 4; ++nt) {
    #pragma unroll
    for (int r = 0; r < 8; ++r) {
      int mEl = r + hv * 8;                      // C/D layout: VGPR r, lane halves
      float g = cg[nt][r], u = cu[nt][r];
      float sv = g * __builtin_amdgcn_rcpf(1.0f + __expf(-g)) * u;
      size_t row = (size_t)e * CAP + m0 + wm + mEl;
      act[row * FDIM + n0 + wn + nt * 16 + l16] = f2bf(sv);
    }
  }
}

// ------------- 8. down GEMM + gather + gate*y + residual --------------------
__global__ __launch_bounds__(256) void moe_gemm2(const unsigned short* __restrict__ act,
                                                 const float* __restrict__ wd,
                                                 const int* __restrict__ slot2tok,
                                                 const float* __restrict__ gate,
                                                 const float* __restrict__ x,
                                                 float* __restrict__ out) {
  __shared__ __align__(16) unsigned short sA[2][64  * 40];
  __shared__ __align__(16) unsigned short sB[2][128 * 40];
  int bx = blockIdx.x;
  int e = bx / 80, rem = bx % 80;
  int m0 = (rem / 8) * 64, n0 = (rem % 8) * 128;
  int tid = threadIdx.x, lane = tid & 31, wave = tid >> 5;
  int wm = (wave & 3) * 16, wn = (wave >> 2) * 64;
  int hv = lane >> 4, l16 = lane & 15;

  v8f cacc[4] = {};

  const unsigned short* Abase = act + ((size_t)e * CAP + m0) * FDIM;
  const float* B = wd + (size_t)e * FDIM * HDIM + n0;
  int arow = tid >> 2, aseg = (tid & 3) * 8;
  int kp = tid >> 4, nc0 = (tid & 15) * 8;

  async_ld_b128(lds_addr(&sA[0][arow * 40 + aseg]),
                Abase + (size_t)arow * FDIM + aseg);
  stage_b(sB[0], B, HDIM, kp, nc0);

  int cur = 0;
  for (int k0 = 0; k0 < FDIM; k0 += 32) {
    wait_async0();
    __syncthreads();
    int nxt = cur ^ 1;
    if (k0 + 32 < FDIM) {
      async_ld_b128(lds_addr(&sA[nxt][arow * 40 + aseg]),
                    Abase + (size_t)arow * FDIM + (k0 + 32) + aseg);
      stage_b(sB[nxt], B + (size_t)(k0 + 32) * HDIM, HDIM, kp, nc0);
      if (k0 + 96 <= FDIM) {
        __builtin_prefetch(B + (size_t)(k0 + 64 + 2 * kp) * HDIM + nc0, 0, 1);
      }
    }
    v16bf a = load_frag(sA[cur], wm + l16, hv);
    v16bf fb[4];
    #pragma unroll
    for (int nt = 0; nt < 4; ++nt)
      fb[nt] = load_frag(sB[cur], wn + nt * 16 + l16, hv);
    #pragma unroll
    for (int nt = 0; nt < 4; ++nt)
      cacc[nt] = __builtin_amdgcn_wmma_f32_16x16x32_bf16(false, a, false, fb[nt],
                                                         (short)0, cacc[nt], false, false);
    cur = nxt;
  }
  // epilogue: gather slot -> token, out = x + gate * y (accepted rows only)
  #pragma unroll
  for (int r = 0; r < 8; ++r) {
    int mEl = r + hv * 8;
    int row = m0 + wm + mEl;
    int t = slot2tok[e * CAP + row];
    if (t >= 0) {
      float gt = gate[t];
      #pragma unroll
      for (int nt = 0; nt < 4; ++nt) {
        size_t idx = (size_t)t * HDIM + n0 + wn + nt * 16 + l16;
        out[idx] = x[idx] + gt * cacc[nt][r];
      }
    }
  }
}

extern "C" void kernel_launch(void* const* d_in, const int* in_sizes, int n_in,
                              void* d_out, int out_size, void* d_ws, size_t ws_size,
                              hipStream_t stream) {
  (void)in_sizes; (void)n_in; (void)out_size; (void)ws_size;
  const float* x  = (const float*)d_in[0];
  const float* wr = (const float*)d_in[1];
  const float* wg = (const float*)d_in[2];
  const float* wu = (const float*)d_in[3];
  const float* wd = (const float*)d_in[4];
  float* out = (float*)d_out;

  char* ws = (char*)d_ws;
  size_t off = 0;
  auto alloc = [&](size_t bytes) -> void* {
    void* p = ws + off;
    off += (bytes + 255) & ~(size_t)255;
    return p;
  };
  int*   top1 = (int*)alloc((size_t)T_TOK * 4);
  float* gate = (float*)alloc((size_t)T_TOK * 4);
  int*   rank = (int*)alloc((size_t)T_TOK * 4);
  int*   cc   = (int*)alloc((size_t)NCHUNK * 32 * 4);
  int*   cb   = (int*)alloc((size_t)NCHUNK * 32 * 4);
  int*   slot = (int*)alloc((size_t)T_TOK * 4);
  int*   s2t  = (int*)alloc((size_t)NEXP * CAP * 4);
  unsigned short* bufbf = (unsigned short*)alloc((size_t)NEXP * CAP * HDIM * 2);
  unsigned short* act   = (unsigned short*)alloc((size_t)NEXP * CAP * FDIM * 2);

  moe_router  <<<T_TOK / 8,        256, 0, stream>>>(x, wr, top1, gate);
  moe_count   <<<NCHUNK / 8,       256, 0, stream>>>(top1, cc, rank);
  moe_scan    <<<1,                 32, 0, stream>>>(cc, cb);
  moe_init_s2t<<<NEXP * CAP / 256, 256, 0, stream>>>(s2t);
  moe_assign  <<<T_TOK / 256,      256, 0, stream>>>(top1, rank, cb, slot, s2t);
  moe_scatter <<<T_TOK,            256, 0, stream>>>(x, slot, bufbf, out);
  moe_gemm1   <<<NEXP * 10 * 8,    256, 0, stream>>>(bufbf, wg, wu, act);
  moe_gemm2   <<<NEXP * 10 * 8,    256, 0, stream>>>(act, wd, s2t, gate, x, out);
}
